// LipschitzNet_8169027797169
// MI455X (gfx1250) — matrix operand order repaired and checked by reading, not compiled, not run on hardware
//
#include <hip/hip_runtime.h>

// ---------------------------------------------------------------------------
// LipschitzRNN on MI455X (gfx1250):
//   A = M_A - 0.5*M_A^T - 0.01*I   (beta=0.75 algebra), same for W.
//   512 sequential Euler steps of h[128,1024]:
//        h += DT*(h@A) + DT*tanh(h@W + x_t^T@E_w^T + E_b)
//   head: h_T @ D_w^T + D_b -> [128,24]
//
// v_wmma_f32_16x16x32_bf16 with fp32 accumulate. fp32 master state shadowed
// by a bf16 copy written each step -> hot K-loop is pure b128 loads + WMMA.
// Weights pre-transposed to bf16 [n][k] (B-operand contiguous); x re-laid-out
// once to bf16 xt[t][d][b] (Z-GEMM A-operand contiguous). W/A bf16 (4MB)
// stay L2-resident across all 512 steps. Wave tile 32x32 = 8 accumulators.
// Epilogue tanh uses the gfx1250 hardware v_tanh_f32 transcendental.
// ---------------------------------------------------------------------------

#define HIDN 1024
#define INPD 128
#define OUTD 24
#define TSTEPS 512
#define DTC 0.001f

typedef __attribute__((ext_vector_type(16))) __bf16         v16bf;
typedef __attribute__((ext_vector_type(8)))  float          v8f;
typedef __attribute__((ext_vector_type(16))) unsigned short v16u;
typedef __attribute__((ext_vector_type(8)))  unsigned short v8u;

union BfCast16 { v16u u; v16bf b; };

__device__ __forceinline__ unsigned short f2bf(float f) {
    unsigned int u = __float_as_uint(f);
    unsigned int r = u + 0x7FFFu + ((u >> 16) & 1u);   // round-to-nearest-even
    return (unsigned short)(r >> 16);
}

// gfx1250 hardware tanh (TRANS32 op). Trailing v_nop satisfies the
// "1 independent op / V_NOP after TRANS before use" co-execution hazard rule,
// since the compiler cannot schedule around an asm block.
__device__ __forceinline__ float htanh(float x) {
    float r;
    asm volatile("v_tanh_f32 %0, %1\n\tv_nop" : "=v"(r) : "v"(x));
    return r;
}

// B-operand fragment: 16 contiguous bf16 at base+off (uniform base, int off)
__device__ __forceinline__ v16bf ld_bfrag(const unsigned short* __restrict__ base, int off) {
    BfCast16 c;
    c.u = *(const v16u*)(base + off);
    return c.b;
}

// A-operand fragment: K chunks [0..7] and [8..15] relative to base+off
__device__ __forceinline__ v16bf ld_afrag(const unsigned short* __restrict__ base, int off) {
    v8u lo = *(const v8u*)(base + off);
    v8u hi = *(const v8u*)(base + off + 8);
    BfCast16 c;
#pragma unroll
    for (int i = 0; i < 8; ++i) { c.u[i] = lo[i]; c.u[i + 8] = hi[i]; }
    return c.b;
}

__device__ __forceinline__ v8f bwmma(v16bf a, v16bf b, v8f c) {
    return __builtin_amdgcn_wmma_f32_16x16x32_bf16(false, a, false, b,
                                                   (short)0, c, false, false);
}

// ---------------------------------------------------------------------------
// One-time preparation kernels
// ---------------------------------------------------------------------------

// Wtb[n*1024+k] = bf16(W[k][n]) with W = M_W - 0.5*M_W^T - 0.01*I; same for A.
__global__ __launch_bounds__(256) void prep_AW(const float* __restrict__ M_W,
                                               const float* __restrict__ M_A,
                                               unsigned short* __restrict__ Wtb,
                                               unsigned short* __restrict__ Atb) {
    int idx = blockIdx.x * 256 + threadIdx.x;          // idx = n*1024 + k
    int k = idx & (HIDN - 1);
    int n = idx >> 10;
    float d = (k == n) ? 0.01f : 0.0f;
    float w = M_W[(size_t)k * HIDN + n] - 0.5f * M_W[(size_t)n * HIDN + k] - d;
    float a = M_A[(size_t)k * HIDN + n] - 0.5f * M_A[(size_t)n * HIDN + k] - d;
    Wtb[idx] = f2bf(w);
    Atb[idx] = f2bf(a);
}

// xt[t][d][b] = bf16(x[b][t][d])
__global__ __launch_bounds__(256) void prep_xt(const float* __restrict__ x,
                                               unsigned short* __restrict__ xt) {
    int idx = blockIdx.x * 256 + threadIdx.x;          // t*16384 + d*128 + b
    int b = idx & (INPD - 1);
    int d = (idx >> 7) & (INPD - 1);
    int t = idx >> 14;
    xt[idx] = f2bf(x[((size_t)b * TSTEPS + t) * INPD + d]);
}

// bf16(E_w); zero h0 (fp32) and hb0 (bf16)
__global__ __launch_bounds__(256) void prep_misc(const float* __restrict__ E_w,
                                                 unsigned short* __restrict__ Ewb,
                                                 float* __restrict__ h0,
                                                 unsigned short* __restrict__ hb0) {
    int idx = blockIdx.x * 256 + threadIdx.x;          // 0 .. 131071
    Ewb[idx] = f2bf(E_w[idx]);
    h0[idx] = 0.0f;
    hb0[idx] = 0;
}

// ---------------------------------------------------------------------------
// One Euler step. Wave computes a 32(M) x 32(N) tile of h_out.
// grid = 32 blocks x 128 threads (4 waves) -> 128 waves = 4 Mtiles x 32 strips
// ---------------------------------------------------------------------------
__global__ __launch_bounds__(128)
void lips_step(const float* __restrict__ h_in, float* __restrict__ h_out,
               const unsigned short* __restrict__ hbf_in,
               unsigned short* __restrict__ hbf_out,
               const unsigned short* __restrict__ Wtb,   // [n][k] bf16
               const unsigned short* __restrict__ Atb,   // [n][k] bf16
               const unsigned short* __restrict__ xt,    // [t][d][b] bf16
               const unsigned short* __restrict__ Ewb,   // [h][b] bf16
               const float* __restrict__ E_b, int t) {
    const int lane  = threadIdx.x & 31;
    const int wave  = threadIdx.x >> 5;
    const int gw    = blockIdx.x * 4 + wave;  // 0..127
    const int mt    = gw & 3;                 // 4 row tiles of 32
    const int ns    = gw >> 2;                // 32 col strips of 32
    const int row0  = mt * 32;
    const int col0  = ns * 32;
    const int lrow  = lane & 15;
    const int khalf = (lane >> 4) * 16;       // A-fragment K sub-offset
    const int kh16  = (lane >> 4) * 16;       // B-fragment K sub-offset

    v8f accW[2][2] = {{{}, {}}, {{}, {}}};
    v8f accA[2][2] = {{{}, {}}, {{}, {}}};

    // ---- Z phase: (x_t^T @ E_w^T), K = batch = 128 ----
    const int xoff0 = (t * INPD + row0 + lrow) * INPD + khalf;        // rows row0..+15
    const int xoff1 = xoff0 + 16 * INPD;                              // rows row0+16..+31
    const int eoff0 = (col0 + lrow) * INPD + kh16;                    // cols col0..+15
    const int eoff1 = eoff0 + 16 * INPD;                              // cols col0+16..+31
#pragma unroll
    for (int kb = 0; kb < 4; ++kb) {
        int kk = kb * 32;
        v16bf xa0 = ld_afrag(xt, xoff0 + kk);
        v16bf xa1 = ld_afrag(xt, xoff1 + kk);
        v16bf e0  = ld_bfrag(Ewb, eoff0 + kk);
        v16bf e1  = ld_bfrag(Ewb, eoff1 + kk);
        accW[0][0] = bwmma(xa0, e0, accW[0][0]);
        accW[0][1] = bwmma(xa0, e1, accW[0][1]);
        accW[1][0] = bwmma(xa1, e0, accW[1][0]);
        accW[1][1] = bwmma(xa1, e1, accW[1][1]);
    }

    // ---- Main phase: h@W and h@A, K = 1024; bf16 shadow state, no cvt ----
    const int hoff0 = (row0 + lrow) * HIDN + khalf;
    const int hoff1 = hoff0 + 16 * HIDN;
    const int coff0 = (col0 + lrow) * HIDN + kh16;
    const int coff1 = coff0 + 16 * HIDN;
#pragma unroll 8
    for (int kb = 0; kb < 32; ++kb) {
        int kk = kb * 32;
        v16bf hf0 = ld_afrag(hbf_in, hoff0 + kk);
        v16bf hf1 = ld_afrag(hbf_in, hoff1 + kk);
        v16bf w0  = ld_bfrag(Wtb, coff0 + kk);
        v16bf w1  = ld_bfrag(Wtb, coff1 + kk);
        v16bf a0  = ld_bfrag(Atb, coff0 + kk);
        v16bf a1  = ld_bfrag(Atb, coff1 + kk);
        accW[0][0] = bwmma(hf0, w0, accW[0][0]);
        accW[0][1] = bwmma(hf0, w1, accW[0][1]);
        accW[1][0] = bwmma(hf1, w0, accW[1][0]);
        accW[1][1] = bwmma(hf1, w1, accW[1][1]);
        accA[0][0] = bwmma(hf0, a0, accA[0][0]);
        accA[0][1] = bwmma(hf0, a1, accA[0][1]);
        accA[1][0] = bwmma(hf1, a0, accA[1][0]);
        accA[1][1] = bwmma(hf1, a1, accA[1][1]);
    }

    // ---- Epilogue: h_new = h + DT*(h@A) + DT*tanh(h@W + Z + E_b) ----
    // C/D layout: element r -> M = r + 8*(lane/16), N = lane%16
    const int mbase = row0 + (lane >> 4) * 8;
#pragma unroll
    for (int ni = 0; ni < 2; ++ni) {
        const int n  = col0 + ni * 16 + lrow;
        const float eb = E_b[n];
#pragma unroll
        for (int mi = 0; mi < 2; ++mi) {
#pragma unroll
            for (int r = 0; r < 8; ++r) {
                const int m  = mbase + mi * 16 + r;
                const int o  = m * HIDN + n;
                float hv = h_in[o] + DTC * accA[mi][ni][r]
                                   + DTC * htanh(accW[mi][ni][r] + eb);
                h_out[o]   = hv;
                hbf_out[o] = f2bf(hv);
            }
        }
    }
}

// ---------------------------------------------------------------------------
// Head: out[d][o] = D_b[o] + sum_k hT[d][k]*D_w[o][k]   (128x24, tiny)
// ---------------------------------------------------------------------------
__global__ __launch_bounds__(256) void lips_head(const float* __restrict__ hT,
                                                 const float* __restrict__ D_w,
                                                 const float* __restrict__ D_b,
                                                 float* __restrict__ out) {
    int idx = blockIdx.x * 256 + threadIdx.x;   // 0 .. 3071
    if (idx >= INPD * OUTD) return;
    int o = idx % OUTD;
    int d = idx / OUTD;
    const float* hrow = hT + (size_t)d * HIDN;
    const float* wrow = D_w + (size_t)o * HIDN;
    float acc = D_b[o];
#pragma unroll 8
    for (int k = 0; k < HIDN; ++k) acc = fmaf(hrow[k], wrow[k], acc);
    out[(size_t)d * OUTD + o] = acc;
}

// ---------------------------------------------------------------------------
extern "C" void kernel_launch(void* const* d_in, const int* in_sizes, int n_in,
                              void* d_out, int out_size, void* d_ws, size_t ws_size,
                              hipStream_t stream) {
    const float* x   = (const float*)d_in[0];
    const float* M_W = (const float*)d_in[1];
    const float* M_A = (const float*)d_in[2];
    const float* E_w = (const float*)d_in[3];
    const float* E_b = (const float*)d_in[4];
    const float* D_w = (const float*)d_in[5];
    const float* D_b = (const float*)d_in[6];
    float* out = (float*)d_out;

    // workspace layout (256B-aligned offsets); total ~21.8 MB
    char* ws = (char*)d_ws;
    float*          h0  = (float*)(ws + 0);                  // 512 KB
    float*          h1  = (float*)(ws + 524288);             // 512 KB
    unsigned short* hb0 = (unsigned short*)(ws + 1048576);   // 256 KB
    unsigned short* hb1 = (unsigned short*)(ws + 1310720);   // 256 KB
    unsigned short* Wtb = (unsigned short*)(ws + 1572864);   // 2 MB
    unsigned short* Atb = (unsigned short*)(ws + 3670016);   // 2 MB
    unsigned short* Ewb = (unsigned short*)(ws + 5767168);   // 256 KB
    unsigned short* xtb = (unsigned short*)(ws + 6029312);   // 16 MB

    prep_AW  <<<(HIDN * HIDN) / 256, 256, 0, stream>>>(M_W, M_A, Wtb, Atb);
    prep_xt  <<<(TSTEPS * INPD * INPD) / 256, 256, 0, stream>>>(x, xtb);
    prep_misc<<<(HIDN * INPD) / 256, 256, 0, stream>>>(E_w, Ewb, h0, hb0);

    const float* hin = h0;
    float* hout = h1;
    const unsigned short* hbin = hb0;
    unsigned short* hbout = hb1;
    for (int t = 0; t < TSTEPS; ++t) {
        lips_step<<<32, 128, 0, stream>>>(hin, hout, hbin, hbout,
                                          Wtb, Atb, xtb, Ewb, E_b, t);
        const float* tf = hout; hout = (float*)hin; hin = tf;
        const unsigned short* tb = hbout; hbout = (unsigned short*)hbin; hbin = tb;
    }
    lips_head<<<12, 256, 0, stream>>>(hin, D_w, D_b, out);
}
